// SwinTransformerBlock_30520037605563
// MI455X (gfx1250) — compile-verified, matrix operand order checked by reading, not run on hardware
//
#include <hip/hip_runtime.h>
#include <hip/hip_bf16.h>
#include <math.h>
#include <stdint.h>

// ---------------- types ----------------
typedef _Float16 v8h  __attribute__((ext_vector_type(8)));
typedef _Float16 v16h __attribute__((ext_vector_type(16)));
typedef float    v8f  __attribute__((ext_vector_type(8)));
typedef unsigned int u32x4 __attribute__((ext_vector_type(4)));
typedef int          i32x4 __attribute__((ext_vector_type(4)));
typedef int          i32x8 __attribute__((ext_vector_type(8)));

// Swin geometry
#define BATCH 32
#define HH 56
#define WW_ 56
#define CC 384
#define HEADS 12
#define WS 7
#define SHIFT 3
#define HD 32
#define NWIN 49           // tokens per window
#define NPAD 64           // padded rows per window
#define NWINDOWS 2048     // B * 8 * 8
#define NTOK 100352       // B * 56 * 56
#define MPAD 131072       // NWINDOWS * NPAD
#define SCALE_F 0.17677669529663687f

constexpr int EPI_F16  = 0;
constexpr int EPI_GELU = 1;
constexpr int EPI_PROJ = 2;
constexpr int EPI_FC2  = 3;

#if __has_builtin(__builtin_amdgcn_tensor_load_to_lds)
#define HAVE_TDM 1
#else
#define HAVE_TDM 0
#endif

// ---- load a 16x32-f16 A/B fragment row from LDS ----
// CDNA5 16-bit A layout: lane (m = l%16, half = l/16) holds halves j=0..15 with
// K(j) = j + 8*half + (j>=8 ? 8 : 0)  => two contiguous 8-half (16B) chunks at
// k = 8*half and k = 16 + 8*half.  B (KxN) uses the identical mapping with the
// lane holding column n = l%16, so a transposed-in-LDS B tile loads the same way.
__device__ __forceinline__ v16h frag_ld(const _Float16* rowp, int half) {
  v8h lo = *(const v8h*)(rowp + 8 * half);
  v8h hi = *(const v8h*)(rowp + 16 + 8 * half);
  return __builtin_shufflevector(lo, hi, 0,1,2,3,4,5,6,7,8,9,10,11,12,13,14,15);
}

__device__ __forceinline__ v8f wmma_f16(v16h a, v16h b, v8f c) {
  return __builtin_amdgcn_wmma_f32_16x16x32_f16(false, a, false, b, (short)0, c, false, false);
}

__device__ __forceinline__ int regid(int g) {         // shift-mask region id
  return g < (HH - WS) ? 0 : (g < (HH - SHIFT) ? 1 : 2);
}

// ---- TDM: issue a 2D f16 tile load (tile0 x tile1 elems, row pitch = stride0
// elems) from global into LDS at lds_addr.  D# per CDNA5 ISA
// (group0: count=1, lds_addr, 57-bit global addr, type=2; group1: data_size=2B,
// tensor_dim0/1 = tile dims, tensor_dim0_stride = row pitch).  Wave-level op.
__device__ __forceinline__ void tdm_load_tile(unsigned lds_addr, const _Float16* gptr,
                                              unsigned long long stride0,
                                              unsigned tile0, unsigned tile1) {
#if HAVE_TDM
  unsigned long long ga = (unsigned long long)(uintptr_t)gptr;
  u32x4 g0;
  g0.x = 1u;                                                   // count=1 (valid user D#)
  g0.y = lds_addr;                                             // LDS byte address
  g0.z = (unsigned)ga;                                         // global_addr[31:0]
  g0.w = (unsigned)((ga >> 32) & 0x1ffffffu) | (2u << 30);     // global_addr[56:32] | type=2
  unsigned dim0 = tile0, dim1 = tile1;                         // tensor == tile footprint
  i32x8 g1;
  g1[0] = (int)(1u << 16);                                     // wg_mask=0, data_size=1 (2B)
  g1[1] = (int)((dim0 & 0xffffu) << 16);                       // tensor_dim0[15:0]
  g1[2] = (int)((dim0 >> 16) & 0xffffu) | (int)((dim1 & 0xffffu) << 16);
  g1[3] = (int)((dim1 >> 16) & 0xffffu) | (int)(tile0 << 16);  // tile_dim0
  g1[4] = (int)tile1;                                          // tile_dim1 (tile_dim2=0)
  g1[5] = (int)(stride0 & 0xffffffffull);                      // tensor_dim0_stride lo
  g1[6] = (int)((stride0 >> 32) & 0xffffull);                  // stride hi (dim1_stride=0)
  g1[7] = 0;
  i32x4 z4 = {0, 0, 0, 0};
#if __clang_major__ >= 23
  i32x8 z8 = {0, 0, 0, 0, 0, 0, 0, 0};
  __builtin_amdgcn_tensor_load_to_lds(g0, g1, z4, z4, z8, 0);
#else
  __builtin_amdgcn_tensor_load_to_lds(g0, g1, z4, z4, 0);
#endif
#else
  (void)lds_addr; (void)gptr; (void)stride0; (void)tile0; (void)tile1;
#endif
}

__device__ __forceinline__ void wait_tensorcnt0() {
#if HAVE_TDM && __has_builtin(__builtin_amdgcn_s_wait_tensorcnt)
  __builtin_amdgcn_s_wait_tensorcnt(0);
#endif
}
__device__ __forceinline__ void wait_tensorcnt2() {
#if HAVE_TDM && __has_builtin(__builtin_amdgcn_s_wait_tensorcnt)
  __builtin_amdgcn_s_wait_tensorcnt(2);
#endif
}

// ---------------- weight fp32 -> f16, transposed to [N][K] ----------------
// one-time tiny cost; makes the W tile TDM-/b128-friendly (same layout as A).
__global__ void transpose_f16_kernel(const float* __restrict__ in, _Float16* __restrict__ out,
                                     int K, int N) {
  int i = blockIdx.x * 256 + threadIdx.x;
  if (i < K * N) {
    int k = i / N, n = i % N;
    out[(size_t)n * K + k] = (_Float16)in[i];
  }
}

// ---------------- LN1 + shift + window partition ----------------
__global__ __launch_bounds__(128)
void ln1_window_kernel(const float* __restrict__ x, const float* __restrict__ g,
                       const float* __restrict__ b, _Float16* __restrict__ win) {
  int pr = blockIdx.x;
  int n = pr & 63, w = pr >> 6;
  _Float16* dst = win + (size_t)pr * CC;
  if (n >= NWIN) {
    for (int i = threadIdx.x; i < CC; i += 128) dst[i] = (_Float16)0.0f;
    return;
  }
  int bb = w >> 6, wy = (w >> 3) & 7, wx = w & 7;
  int iy = n / WS, ix = n % WS;
  int hh = (wy * WS + iy + SHIFT) % HH;
  int ww = (wx * WS + ix + SHIFT) % WW_;
  const float* src = x + ((size_t)bb * (HH * WW_) + hh * WW_ + ww) * CC;

  float v0 = src[threadIdx.x], v1 = src[threadIdx.x + 128], v2 = src[threadIdx.x + 256];
  float s = v0 + v1 + v2;
  #pragma unroll
  for (int m = 16; m; m >>= 1) s += __shfl_xor(s, m, 32);
  __shared__ float part[4];
  __shared__ float stats[2];
  int wv = threadIdx.x >> 5, lane = threadIdx.x & 31;
  if (lane == 0) part[wv] = s;
  __syncthreads();
  if (threadIdx.x == 0) stats[0] = (part[0] + part[1] + part[2] + part[3]) * (1.0f / CC);
  __syncthreads();
  float mu = stats[0];
  float d0 = v0 - mu, d1 = v1 - mu, d2 = v2 - mu;
  float q = d0 * d0 + d1 * d1 + d2 * d2;
  #pragma unroll
  for (int m = 16; m; m >>= 1) q += __shfl_xor(q, m, 32);
  __syncthreads();
  if (lane == 0) part[wv] = q;
  __syncthreads();
  if (threadIdx.x == 0)
    stats[1] = rsqrtf((part[0] + part[1] + part[2] + part[3]) * (1.0f / CC) + 1e-5f);
  __syncthreads();
  float rs = stats[1];
  dst[threadIdx.x]       = (_Float16)(d0 * rs * g[threadIdx.x]       + b[threadIdx.x]);
  dst[threadIdx.x + 128] = (_Float16)(d1 * rs * g[threadIdx.x + 128] + b[threadIdx.x + 128]);
  dst[threadIdx.x + 256] = (_Float16)(d2 * rs * g[threadIdx.x + 256] + b[threadIdx.x + 256]);
}

// ---------------- LN2 ----------------
__global__ __launch_bounds__(128)
void ln2_kernel(const float* __restrict__ x, const float* __restrict__ g,
                const float* __restrict__ b, _Float16* __restrict__ out) {
  int row = blockIdx.x;
  const float* src = x + (size_t)row * CC;
  _Float16* dst = out + (size_t)row * CC;
  float v0 = src[threadIdx.x], v1 = src[threadIdx.x + 128], v2 = src[threadIdx.x + 256];
  float s = v0 + v1 + v2;
  #pragma unroll
  for (int m = 16; m; m >>= 1) s += __shfl_xor(s, m, 32);
  __shared__ float part[4];
  __shared__ float stats[2];
  int wv = threadIdx.x >> 5, lane = threadIdx.x & 31;
  if (lane == 0) part[wv] = s;
  __syncthreads();
  if (threadIdx.x == 0) stats[0] = (part[0] + part[1] + part[2] + part[3]) * (1.0f / CC);
  __syncthreads();
  float mu = stats[0];
  float d0 = v0 - mu, d1 = v1 - mu, d2 = v2 - mu;
  float q = d0 * d0 + d1 * d1 + d2 * d2;
  #pragma unroll
  for (int m = 16; m; m >>= 1) q += __shfl_xor(q, m, 32);
  __syncthreads();
  if (lane == 0) part[wv] = q;
  __syncthreads();
  if (threadIdx.x == 0)
    stats[1] = rsqrtf((part[0] + part[1] + part[2] + part[3]) * (1.0f / CC) + 1e-5f);
  __syncthreads();
  float rs = stats[1];
  dst[threadIdx.x]       = (_Float16)(d0 * rs * g[threadIdx.x]       + b[threadIdx.x]);
  dst[threadIdx.x + 128] = (_Float16)(d1 * rs * g[threadIdx.x + 128] + b[threadIdx.x + 128]);
  dst[threadIdx.x + 256] = (_Float16)(d2 * rs * g[threadIdx.x + 256] + b[threadIdx.x + 256]);
}

// ------- generic WMMA GEMM: out(M x ND) = A(M x KD) @ Wt(ND x KD)^T + bias -------
// block = 256 thr = 8 waves; block tile 128(M) x 128(N); wave tile 32(M) x 64(N).
// Both A and W tiles are DMA'd by the Tensor Data Mover into double-buffered LDS:
// wave 0 issues tile i+1 while everyone computes tile i; "s_wait_tensorcnt 2"
// drains exactly the current tile's two descriptors.  8 WMMA / wave / K-step.
template<int KD, int ND, int EPI>
__global__ __launch_bounds__(256)
void wmma_gemm_kernel(const _Float16* __restrict__ A, const _Float16* __restrict__ Wt,
                      const float* __restrict__ bias, void* __restrict__ out,
                      const float* __restrict__ addend) {
  __shared__ _Float16 As[2][128 * 32];
  __shared__ _Float16 Ws[2][128 * 32];   // [n][k] (Wt is pre-transposed in global)
  int t = threadIdx.x;
  int bm = blockIdx.y, bn = blockIdx.x;
  int w = t >> 5, l = t & 31, half = l >> 4, ln = l & 15;
  int wm = w & 3, wn = w >> 2;
  constexpr int NK = KD / 32;

  v8f acc[2][4] = {};

  const _Float16* Atile = A  + (size_t)(bm * 128) * KD;
  const _Float16* Wtile = Wt + (size_t)(bn * 128) * KD;
#if HAVE_TDM
  unsigned as_lds[2] = { (unsigned)(uintptr_t)(void*)&As[0][0],
                         (unsigned)(uintptr_t)(void*)&As[1][0] };
  unsigned ws_lds[2] = { (unsigned)(uintptr_t)(void*)&Ws[0][0],
                         (unsigned)(uintptr_t)(void*)&Ws[1][0] };
  if (w == 0) {   // prologue: tile 0 -> buffer 0
    tdm_load_tile(as_lds[0], Atile, (unsigned long long)KD, 32u, 128u);
    tdm_load_tile(ws_lds[0], Wtile, (unsigned long long)KD, 32u, 128u);
  }
#else
  int srow = t & 127, sc0 = (t >> 7) * 16;
  const _Float16* Aim = Atile + (size_t)srow * KD + sc0;
  const _Float16* Wim = Wtile + (size_t)srow * KD + sc0;
#endif

  #pragma unroll 2
  for (int kstep = 0; kstep < NK; kstep++) {
    int cur = kstep & 1;
#if HAVE_TDM
    if (kstep + 1 < NK) {
      if (w == 0) {   // issue tile i+1 into the buffer freed by iteration i-1
        tdm_load_tile(as_lds[cur ^ 1], Atile + (kstep + 1) * 32, (unsigned long long)KD, 32u, 128u);
        tdm_load_tile(ws_lds[cur ^ 1], Wtile + (kstep + 1) * 32, (unsigned long long)KD, 32u, 128u);
      }
      wait_tensorcnt2();        // drain tile i (its 2 descriptors are oldest in flight)
    } else {
      wait_tensorcnt0();
    }
#else
    *(v8h*)&As[cur][srow * 32 + sc0]     = *(const v8h*)(Aim + kstep * 32);
    *(v8h*)&As[cur][srow * 32 + sc0 + 8] = *(const v8h*)(Aim + kstep * 32 + 8);
    *(v8h*)&Ws[cur][srow * 32 + sc0]     = *(const v8h*)(Wim + kstep * 32);
    *(v8h*)&Ws[cur][srow * 32 + sc0 + 8] = *(const v8h*)(Wim + kstep * 32 + 8);
#endif
    __syncthreads();

    v16h a0 = frag_ld(&As[cur][(wm * 32 + ln) * 32], half);
    v16h a1 = frag_ld(&As[cur][(wm * 32 + 16 + ln) * 32], half);
    #pragma unroll
    for (int s = 0; s < 4; s++) {
      v16h b = frag_ld(&Ws[cur][(wn * 64 + s * 16 + ln) * 32], half);
      acc[0][s] = wmma_f16(a0, b, acc[0][s]);
      acc[1][s] = wmma_f16(a1, b, acc[1][s]);
    }
    __syncthreads();            // frees buffer cur for the tile issued next iteration
  }

  // ---- epilogue ----
  #pragma unroll
  for (int u = 0; u < 2; u++) {
    #pragma unroll
    for (int s = 0; s < 4; s++) {
      int col = bn * 128 + wn * 64 + s * 16 + ln;
      float bv = bias[col];
      #pragma unroll
      for (int r = 0; r < 8; r++) {
        int m = wm * 32 + u * 16 + r + 8 * half;
        float v = acc[u][s][r] + bv;
        if (EPI == EPI_F16) {
          ((_Float16*)out)[(size_t)(bm * 128 + m) * ND + col] = (_Float16)v;
        } else if (EPI == EPI_GELU) {
          v = 0.5f * v * (1.0f + erff(v * 0.70710678118654752f));
          ((_Float16*)out)[(size_t)(bm * 128 + m) * ND + col] = (_Float16)v;
        } else if (EPI == EPI_PROJ) {
          int pr = bm * 128 + m;              // padded row
          int n = pr & 63;
          if (n < NWIN) {
            int wi = pr >> 6;
            int bb = wi >> 6, wy = (wi >> 3) & 7, wx = wi & 7;
            int iy = n / WS, ix = n % WS;
            int hh = (wy * WS + iy + SHIFT) % HH;
            int ww = (wx * WS + ix + SHIFT) % WW_;
            size_t tok = (size_t)bb * (HH * WW_) + hh * WW_ + ww;
            ((float*)out)[tok * CC + col] = addend[tok * CC + col] + v;
          }
        } else { // EPI_FC2
          size_t row = (size_t)(bm * 128 + m);
          ((float*)out)[row * CC + col] = v + addend[row * CC + col];
        }
      }
    }
  }
}

// ---------------- fused window attention ----------------
// one block (128 thr = 4 waves) per (window, head); padded 64x64 score tile.
// Q/K tiles (64x32, pitch 1152) DMA'd by TDM; V^T staged by VALU (needs transpose).
__global__ __launch_bounds__(128)
void attn_kernel(const _Float16* __restrict__ qkv, const float* __restrict__ rpb,
                 _Float16* __restrict__ out) {
  int wh = blockIdx.x;
  int h = wh % HEADS, win = wh / HEADS;
  int wy = (win >> 3) & 7, wx = win & 7;

  __shared__ _Float16 qs[64 * 32];
  __shared__ _Float16 ks[64 * 32];
  __shared__ _Float16 vt[32 * 64];   // V transposed: vt[d][k]
  __shared__ _Float16 ps[64 * 64];   // softmax(P) in f16
  __shared__ float rpbs[(2 * WS - 1) * (2 * WS - 1)];  // 169, this head only

  int t = threadIdx.x;
  const _Float16* qbase = qkv + (size_t)(win * 64) * (3 * CC) + h * HD;
#if HAVE_TDM
  if (t < 32) {   // wave 0 issues both DMA tiles
    tdm_load_tile((unsigned)(uintptr_t)(void*)&qs[0], qbase,          3 * CC, 32u, 64u);
    tdm_load_tile((unsigned)(uintptr_t)(void*)&ks[0], qbase + CC,     3 * CC, 32u, 64u);
  }
#endif
  for (int i = t; i < 169; i += 128) rpbs[i] = rpb[i * HEADS + h];

  { // stage V transposed (and Q/K too when TDM is unavailable)
    int row = t >> 1, c0 = (t & 1) * 16;
    const _Float16* base = qbase + (size_t)row * (3 * CC) + c0;
#if !HAVE_TDM
    const v8h* qp = (const v8h*)(base);
    const v8h* kp = (const v8h*)(base + CC);
    *(v8h*)&qs[row * 32 + c0]     = qp[0];
    *(v8h*)&qs[row * 32 + c0 + 8] = qp[1];
    *(v8h*)&ks[row * 32 + c0]     = kp[0];
    *(v8h*)&ks[row * 32 + c0 + 8] = kp[1];
#endif
    const v8h* vp = (const v8h*)(base + 2 * CC);
    v8h va = vp[0], vb = vp[1];
    #pragma unroll
    for (int j = 0; j < 8; j++) {
      vt[(c0 + j) * 64 + row]     = va[j];
      vt[(c0 + 8 + j) * 64 + row] = vb[j];
    }
  }
  wait_tensorcnt0();
  __syncthreads();

  int w = t >> 5, l = t & 31, half = l >> 4, ln = l & 15;

  // S = Q @ K^T  (B fragment of K^T == K rows loaded A-style)
  v8f sacc[4] = {};
  {
    v16h a = frag_ld(&qs[(w * 16 + ln) * 32], half);
    #pragma unroll
    for (int s = 0; s < 4; s++) {
      v16h b = frag_ld(&ks[(s * 16 + ln) * 32], half);
      sacc[s] = wmma_f16(a, b, sacc[s]);
    }
  }

  // bias + shift-mask + pad-mask + softmax
  #pragma unroll
  for (int r = 0; r < 8; r++) {
    int i = w * 16 + r + 8 * half;
    int iy = i / WS, ix = i % WS;
    int ri = 0;
    if (i < NWIN) ri = 3 * regid(wy * WS + iy) + regid(wx * WS + ix);
    float e[4];
    float mx = -3.0e38f;
    #pragma unroll
    for (int s = 0; s < 4; s++) {
      int j = ln + 16 * s;
      float v = sacc[s][r] * SCALE_F;
      if (j < NWIN && i < NWIN) {
        int jy = j / WS, jx = j % WS;
        v += rpbs[(iy - jy + WS - 1) * (2 * WS - 1) + (ix - jx + WS - 1)];
        int rj = 3 * regid(wy * WS + jy) + regid(wx * WS + jx);
        if (rj != ri) v -= 100.0f;
      } else if (j >= NWIN) {
        v = -1.0e30f;
      }
      e[s] = v;
      mx = fmaxf(mx, v);
    }
    #pragma unroll
    for (int m = 1; m < 16; m <<= 1) mx = fmaxf(mx, __shfl_xor(mx, m, 32));
    float sum = 0.0f;
    #pragma unroll
    for (int s = 0; s < 4; s++) { e[s] = __expf(e[s] - mx); sum += e[s]; }
    #pragma unroll
    for (int m = 1; m < 16; m <<= 1) sum += __shfl_xor(sum, m, 32);
    float inv = 1.0f / sum;
    #pragma unroll
    for (int s = 0; s < 4; s++) ps[i * 64 + ln + 16 * s] = (_Float16)(e[s] * inv);
  }
  __syncthreads();

  // O = P @ V  (B fragment from vt[d][k])
  v8f oacc[2] = {};
  #pragma unroll
  for (int kq = 0; kq < 2; kq++) {
    v16h a = frag_ld(&ps[(w * 16 + ln) * 64 + kq * 32], half);
    #pragma unroll
    for (int s = 0; s < 2; s++) {
      v16h b = frag_ld(&vt[(s * 16 + ln) * 64 + kq * 32], half);
      oacc[s] = wmma_f16(a, b, oacc[s]);
    }
  }
  #pragma unroll
  for (int s = 0; s < 2; s++) {
    int d = ln + 16 * s;
    #pragma unroll
    for (int r = 0; r < 8; r++) {
      int i = w * 16 + r + 8 * half;
      out[(size_t)(win * 64 + i) * CC + h * HD + d] = (_Float16)oacc[s][r];
    }
  }
}

// ---------------- launcher ----------------
extern "C" void kernel_launch(void* const* d_in, const int* in_sizes, int n_in,
                              void* d_out, int out_size, void* d_ws, size_t ws_size,
                              hipStream_t stream) {
  const float* x      = (const float*)d_in[0];
  const float* ln1_g  = (const float*)d_in[1];
  const float* ln1_b  = (const float*)d_in[2];
  const float* qkv_w  = (const float*)d_in[3];
  const float* qkv_b  = (const float*)d_in[4];
  const float* rpb    = (const float*)d_in[5];
  const float* proj_w = (const float*)d_in[6];
  const float* proj_b = (const float*)d_in[7];
  const float* ln2_g  = (const float*)d_in[8];
  const float* ln2_b  = (const float*)d_in[9];
  const float* fc1_w  = (const float*)d_in[10];
  const float* fc1_b  = (const float*)d_in[11];
  const float* fc2_w  = (const float*)d_in[12];
  const float* fc2_b  = (const float*)d_in[13];
  float* out = (float*)d_out;
  (void)in_sizes; (void)n_in; (void)out_size; (void)ws_size;

  char* ws = (char*)d_ws;
  _Float16* wqkv  = (_Float16*)(ws + 0);                       // 1152 x 384 f16 (transposed)
  _Float16* wproj = (_Float16*)(ws + 884736ull);               // 384 x 384
  _Float16* wfc1  = (_Float16*)(ws + 1179648ull);              // 1536 x 384
  _Float16* wfc2  = (_Float16*)(ws + 2359296ull);              // 384 x 1536
  _Float16* win   = (_Float16*)(ws + 4194304ull);              // 131072 x 384 f16
  _Float16* qkvo  = (_Float16*)(ws + 104857600ull);            // 131072 x 1152 f16
  _Float16* atto  = (_Float16*)(ws + 406847488ull);            // 131072 x 384 f16
  float*    xres  = (float*)  (ws + 507510784ull);             // 100352 x 384 f32
  _Float16* ln2o  = win;    // reuse (win dead after QKV GEMM)
  _Float16* h1    = qkvo;   // reuse (qkv dead after attention)

  transpose_f16_kernel<<<(384 * 1152 + 255) / 256, 256, 0, stream>>>(qkv_w, wqkv, 384, 1152);
  transpose_f16_kernel<<<(384 * 384  + 255) / 256, 256, 0, stream>>>(proj_w, wproj, 384, 384);
  transpose_f16_kernel<<<(384 * 1536 + 255) / 256, 256, 0, stream>>>(fc1_w, wfc1, 384, 1536);
  transpose_f16_kernel<<<(1536 * 384 + 255) / 256, 256, 0, stream>>>(fc2_w, wfc2, 1536, 384);

  ln1_window_kernel<<<MPAD, 128, 0, stream>>>(x, ln1_g, ln1_b, win);

  wmma_gemm_kernel<384, 1152, EPI_F16><<<dim3(9, 1024), 256, 0, stream>>>(
      win, wqkv, qkv_b, (void*)qkvo, nullptr);

  attn_kernel<<<NWINDOWS * HEADS, 128, 0, stream>>>(qkvo, rpb, atto);

  wmma_gemm_kernel<384, 384, EPI_PROJ><<<dim3(3, 1024), 256, 0, stream>>>(
      atto, wproj, proj_b, (void*)xres, x);

  ln2_kernel<<<NTOK, 128, 0, stream>>>(xres, ln2_g, ln2_b, ln2o);

  wmma_gemm_kernel<384, 1536, EPI_GELU><<<dim3(12, 784), 256, 0, stream>>>(
      ln2o, wfc1, fc1_b, (void*)h1, nullptr);

  wmma_gemm_kernel<1536, 384, EPI_FC2><<<dim3(3, 784), 256, 0, stream>>>(
      h1, wfc2, fc2_b, (void*)out, xres);
}
